// MHAHead_50388556317359
// MI455X (gfx1250) — matrix operand order repaired and checked by reading, not compiled, not run on hardware
//
#include <hip/hip_runtime.h>
#include <hip/hip_bf16.h>

// MHA single-head attention for MI455X (gfx1250), wave32 + WMMA bf16 + TDM.
// Kernel 1: QKV projection (fp32 in -> bf16 Q/K row-major, V transposed [B,E,S]).
// Kernel 2: flash attention, 64 queries / block, online softmax, fp32 accum.
//           K/V chunks staged into LDS by the Tensor Data Mover, double-buffered.

#define Bb 8
#define Ss 4096
#define Ee 256

typedef __attribute__((ext_vector_type(16))) __bf16 v16bf;
typedef __attribute__((ext_vector_type(8)))  __bf16 v8bf;
typedef __attribute__((ext_vector_type(8)))  float  v8f;
typedef __attribute__((ext_vector_type(4)))  unsigned int v4u;
typedef __attribute__((ext_vector_type(8)))  int v8i;
typedef __attribute__((ext_vector_type(4)))  int v4i;

// Generic pointer to LDS: low 32 bits are the wave-relative LDS byte offset.
static __device__ inline unsigned lds_off(const void* p) {
    return (unsigned)(unsigned long long)(uintptr_t)p;
}

// Issue a 2-D TDM load: tile_d1 rows of tile_d0 bf16 elements, row stride
// stride0 (elements), global -> LDS (contiguous row-major fill).
static __device__ inline void tdm_load_2d(unsigned lds_addr, const void* gaddr,
                                          unsigned tensor_d0, unsigned tensor_d1,
                                          unsigned tile_d0, unsigned tile_d1,
                                          unsigned long long stride0)
{
    const unsigned long long ga = (unsigned long long)(uintptr_t)gaddr;
    v4u g0;
    g0[0] = 1u;                                           // count=1 (valid D#)
    g0[1] = lds_addr;                                     // LDS byte address
    g0[2] = (unsigned)(ga & 0xFFFFFFFFu);                 // global_addr[31:0]
    g0[3] = (unsigned)((ga >> 32) & 0x01FFFFFFu)          // global_addr[56:32]
          | (2u << 30);                                   // type = 2 ("image")
    v8i g1;
    g1[0] = (int)(1u << 16);                              // data_size = 2 bytes
    g1[1] = (int)((tensor_d0 & 0xFFFFu) << 16);           // tensor_dim0[15:0]
    g1[2] = (int)((tensor_d0 >> 16) |                     // tensor_dim0[31:16]
                  ((tensor_d1 & 0xFFFFu) << 16));         // tensor_dim1[15:0]
    g1[3] = (int)((tensor_d1 >> 16) | (tile_d0 << 16));   // tile_dim0
    g1[4] = (int)(tile_d1 & 0xFFFFu);                     // tile_dim1, tile_dim2=0
    g1[5] = (int)(stride0 & 0xFFFFFFFFu);                 // tensor_dim0_stride lo
    g1[6] = (int)((stride0 >> 32) & 0xFFFFu);             // stride hi, dim1_stride=0
    g1[7] = 0;
    v4i g2 = {0, 0, 0, 0};                                // dims 2/3 unused (2-D)
    v4i g3 = {0, 0, 0, 0};                                // dim 4 unused
    v8i g4 = {0, 0, 0, 0, 0, 0, 0, 0};                    // unused trailing group
    __builtin_amdgcn_tensor_load_to_lds(g0, g1, g2, g3, g4, 0);
}

// ---------------------------------------------------------------------------
// Kernel 1: q = x @ Wq^T + bq (same for k, v). One 16x16 output tile per wave.
// ---------------------------------------------------------------------------
__global__ __launch_bounds__(256) void qkv_proj_kernel(
    const float* __restrict__ x,
    const float* __restrict__ Wq, const float* __restrict__ bq,
    const float* __restrict__ Wk, const float* __restrict__ bk,
    const float* __restrict__ Wv, const float* __restrict__ bv,
    __bf16* __restrict__ Qb, __bf16* __restrict__ Kb, __bf16* __restrict__ Vt)
{
    const int lane = threadIdx.x & 31;
    const int wave = threadIdx.x >> 5;
    const int tile = blockIdx.x * 8 + wave;

    const int mat = tile >> 15;          // 0:q 1:k 2:v
    const int rem = tile & 32767;
    const int b   = rem >> 12;           // batch
    const int rt  = (rem & 4095) >> 4;   // row tile (sequence), 0..255
    const int ct  = rem & 15;            // col tile (embed), 0..15

    const float* W    = (mat == 0) ? Wq : (mat == 1) ? Wk : Wv;
    const float* bias = (mat == 0) ? bq : (mat == 1) ? bk : bv;

    const int n    = lane & 15;
    const int half = lane >> 4;
    const int kbA  = half * 8;
    const int ksB  = half * 16;

    const float* xrow = x + ((size_t)b * Ss + rt * 16 + n) * Ee;
    const float* wrow = W + (size_t)(ct * 16 + n) * Ee;

    v8f acc = {};
    #pragma unroll
    for (int d0 = 0; d0 < Ee; d0 += 32) {
        v16bf a, bm;
        #pragma unroll
        for (int e = 0; e < 8; ++e) {
            a[e]     = (__bf16)xrow[d0 + kbA + e];
            a[e + 8] = (__bf16)xrow[d0 + 16 + kbA + e];
        }
        #pragma unroll
        for (int e = 0; e < 8; ++e) {
            bm[e]     = (__bf16)wrow[d0 + kbA + e];
            bm[e + 8] = (__bf16)wrow[d0 + 16 + kbA + e];
        }
        acc = __builtin_amdgcn_wmma_f32_16x16x32_bf16(
            false, a, false, bm, (short)0, acc, false, false);
    }

    const float bc = bias[ct * 16 + n];
    if (mat == 2) {
        __bf16* vo = Vt + ((size_t)b * Ee + ct * 16 + n) * Ss + rt * 16 + half * 8;
        #pragma unroll
        for (int i = 0; i < 8; ++i) vo[i] = (__bf16)(acc[i] + bc);
    } else {
        __bf16* o = (mat == 0 ? Qb : Kb)
                  + ((size_t)b * Ss + rt * 16 + half * 8) * Ee + ct * 16 + n;
        #pragma unroll
        for (int i = 0; i < 8; ++i) o[(size_t)i * Ee] = (__bf16)(acc[i] + bc);
    }
}

// ---------------------------------------------------------------------------
// Kernel 2: flash attention with TDM double-buffered K/V staging.
// ---------------------------------------------------------------------------
__global__ __launch_bounds__(256) void flash_attn_kernel(
    const __bf16* __restrict__ Qb, const __bf16* __restrict__ Kb,
    const __bf16* __restrict__ Vt, float* __restrict__ out)
{
    __shared__ __bf16 Kbuf[2][32 * Ee];   // 2 x 16 KB: 32 keys x 256 dims
    __shared__ __bf16 Vbuf[2][Ee * 32];   // 2 x 16 KB: 256 dims x 32 keys (Vt)
    __shared__ float  Sbuf[64 * 32];
    __shared__ __bf16 Pbuf[64 * 32];
    __shared__ float  Mrow[64], Lrow[64], Arow[64];

    const int lane = threadIdx.x & 31;
    const int wave = threadIdx.x >> 5;
    const int r    = wave & 3;         // query row tile 0..3
    const int cg   = wave >> 2;        // 0..1 : key half / output col half
    const int n    = lane & 15;
    const int half = lane >> 4;
    const int kbA  = half * 8;
    const int ksB  = half * 16;

    const int b  = blockIdx.x >> 6;
    const int q0 = (blockIdx.x & 63) * 64;

    const __bf16* Kbase  = Kb + (size_t)b * Ss * Ee;   // [S][E]
    const __bf16* Vtbase = Vt + (size_t)b * Ee * Ss;   // [E][S]

    // Preload chunk 0 into buffer 0 (wave 0 drives the TDM).
    if (wave == 0) {
        tdm_load_2d(lds_off(&Kbuf[0][0]), Kbase, Ee, 32, Ee, 32, Ee);
        tdm_load_2d(lds_off(&Vbuf[0][0]), Vtbase, 32, Ee, 32, Ee, Ss);
    }

    // Pin Q A-fragments in registers: rows q0+16r+m, full E=256 (8 k-steps).
    v16bf qf[8];
    const __bf16* qrow = Qb + ((size_t)b * Ss + q0 + 16 * r + n) * Ee;
    #pragma unroll
    for (int kk = 0; kk < 8; ++kk) {
        v8bf lo = *(const v8bf*)(qrow + kk * 32 + kbA);
        v8bf hi = *(const v8bf*)(qrow + kk * 32 + 16 + kbA);
        qf[kk] = __builtin_shufflevector(lo, hi,
                 0,1,2,3,4,5,6,7,8,9,10,11,12,13,14,15);
    }

    v8f zero = {};
    v8f oacc[8];
    #pragma unroll
    for (int t = 0; t < 8; ++t) oacc[t] = zero;

    if (threadIdx.x < 64) { Mrow[threadIdx.x] = -1e30f; Lrow[threadIdx.x] = 0.0f; }

    for (int kc = 0; kc < Ss / 32; ++kc) {
        const int buf = kc & 1;
        // Current buffer's DMA complete (issuing wave), then make visible.
        if (wave == 0) __builtin_amdgcn_s_wait_tensorcnt((short)0);
        __syncthreads();
        // Kick off next chunk into the other buffer; overlaps with compute.
        if (wave == 0 && kc + 1 < Ss / 32) {
            const int key1 = (kc + 1) * 32;
            tdm_load_2d(lds_off(&Kbuf[buf ^ 1][0]), Kbase + (size_t)key1 * Ee,
                        Ee, 32, Ee, 32, Ee);
            tdm_load_2d(lds_off(&Vbuf[buf ^ 1][0]), Vtbase + key1,
                        32, Ee, 32, Ee, Ss);
        }

        // ---- scores: 16x16 tile = Q[rows] @ K[keys]^T over E=256 ----
        v8f sc = zero;
        const __bf16* krow = &Kbuf[buf][(16 * cg + n) * Ee + kbA];
        #pragma unroll
        for (int kk = 0; kk < 8; ++kk) {
            v8bf klo = *(const v8bf*)(krow + kk * 32);
            v8bf khi = *(const v8bf*)(krow + kk * 32 + 16);
            v16bf kf = __builtin_shufflevector(klo, khi,
                       0,1,2,3,4,5,6,7,8,9,10,11,12,13,14,15);
            sc = __builtin_amdgcn_wmma_f32_16x16x32_bf16(
                false, qf[kk], false, kf, (short)0, sc, false, false);
        }
        #pragma unroll
        for (int i = 0; i < 8; ++i)
            Sbuf[(16 * r + 8 * half + i) * 32 + 16 * cg + n] = sc[i] * 0.0625f;
        __syncthreads();

        // ---- online softmax update, one thread per query row ----
        if (threadIdx.x < 64) {
            const int row = threadIdx.x;
            const float mold = Mrow[row];
            float mc = mold;
            float s[32];
            #pragma unroll
            for (int j = 0; j < 32; ++j) {
                s[j] = Sbuf[row * 32 + j];
                mc = fmaxf(mc, s[j]);
            }
            const float alpha = __expf(mold - mc);
            float ps = 0.0f;
            #pragma unroll
            for (int j = 0; j < 32; ++j) {
                float p = __expf(s[j] - mc);
                ps += p;
                Pbuf[row * 32 + j] = (__bf16)p;
            }
            Mrow[row] = mc;
            Lrow[row] = Lrow[row] * alpha + ps;
            Arow[row] = alpha;
        }
        __syncthreads();

        // ---- rescale O and accumulate P @ V (V from LDS) ----
        const __bf16* prow = &Pbuf[(16 * r + n) * 32];
        v8bf plo = *(const v8bf*)(prow + kbA);
        v8bf phi = *(const v8bf*)(prow + 16 + kbA);
        v16bf pf = __builtin_shufflevector(plo, phi,
                   0,1,2,3,4,5,6,7,8,9,10,11,12,13,14,15);

        float alpha_i[8];
        #pragma unroll
        for (int i = 0; i < 8; ++i) alpha_i[i] = Arow[16 * r + 8 * half + i];

        #pragma unroll
        for (int t = 0; t < 8; ++t) {
            v8f o = oacc[t];
            #pragma unroll
            for (int i = 0; i < 8; ++i) o[i] *= alpha_i[i];
            const __bf16* vrow = &Vbuf[buf][(128 * cg + 16 * t + n) * 32 + kbA];
            v8bf vlo = *(const v8bf*)(vrow);
            v8bf vhi = *(const v8bf*)(vrow + 16);
            v16bf vf = __builtin_shufflevector(vlo, vhi,
                       0,1,2,3,4,5,6,7,8,9,10,11,12,13,14,15);
            oacc[t] = __builtin_amdgcn_wmma_f32_16x16x32_bf16(
                false, pf, false, vf, (short)0, o, false, false);
        }
        __syncthreads();
    }

    // ---- finalize: divide by row sums, store fp32 output ----
    float linv[8];
    #pragma unroll
    for (int i = 0; i < 8; ++i) linv[i] = 1.0f / Lrow[16 * r + 8 * half + i];
    #pragma unroll
    for (int t = 0; t < 8; ++t) {
        float* orow = out + ((size_t)b * Ss + q0 + 16 * r + 8 * half) * Ee
                          + 128 * cg + 16 * t + n;
        #pragma unroll
        for (int i = 0; i < 8; ++i)
            orow[(size_t)i * Ee] = oacc[t][i] * linv[i];
    }
}

// ---------------------------------------------------------------------------
extern "C" void kernel_launch(void* const* d_in, const int* in_sizes, int n_in,
                              void* d_out, int out_size, void* d_ws, size_t ws_size,
                              hipStream_t stream) {
    const float* x  = (const float*)d_in[0];
    const float* Wq = (const float*)d_in[1];
    const float* bq = (const float*)d_in[2];
    const float* Wk = (const float*)d_in[3];
    const float* bk = (const float*)d_in[4];
    const float* Wv = (const float*)d_in[5];
    const float* bv = (const float*)d_in[6];
    float* out = (float*)d_out;

    const size_t elems = (size_t)Bb * Ss * Ee;          // 8M elements
    __bf16* Qb = (__bf16*)d_ws;                         // 16 MB
    __bf16* Kb = (__bf16*)((char*)d_ws + elems * 2);    // 16 MB
    __bf16* Vt = (__bf16*)((char*)d_ws + elems * 4);    // 16 MB (transposed V)

    qkv_proj_kernel<<<12288, 256, 0, stream>>>(x, Wq, bq, Wk, bk, Wv, bv,
                                               Qb, Kb, Vt);
    flash_attn_kernel<<<Bb * (Ss / 64), 256, 0, stream>>>(Qb, Kb, Vt, out);
}
